// GroupQueryAttention_15144054686317
// MI455X (gfx1250) — matrix operand order repaired and checked by reading, not compiled, not run on hardware
//
#include <hip/hip_runtime.h>
#include <hip/hip_bf16.h>
#include <math.h>

// ---------------------------------------------------------------------------
// CDNA5 (gfx1250) transformer block: GQA attention + FFN, bf16 WMMA pipeline.
//   wave32, v_wmma_f32_16x16x32_bf16 for all matmuls, flash-style attention,
//   async global->LDS staging (ASYNCcnt) with double-buffered GEMM tiles.
// ---------------------------------------------------------------------------

typedef __bf16 bf16_t;
typedef __attribute__((ext_vector_type(16))) __bf16 v16bf;
typedef __attribute__((ext_vector_type(8)))  float  v8f;

#define WMMA_BF16(a, b, c) \
  __builtin_amdgcn_wmma_f32_16x16x32_bf16(false, (a), false, (b), (short)0, (c), false, false)

// Async global->LDS copy path (gfx1250 ASYNCcnt engine), guarded so both the
// host pass and older toolchains fall back to the synchronous uint4 path.
// Probe-confirmed signature: (int4* src_global, int4* dst_lds, imm off, imm cpol)
// with GENERIC (vector_size(16) int) pointers — backend infers address spaces.
#if defined(__HIP_DEVICE_COMPILE__) &&                                \
    __has_builtin(__builtin_amdgcn_global_load_async_to_lds_b128) && \
    __has_builtin(__builtin_amdgcn_s_wait_asynccnt)
#define USE_ASYNC_COPY 1
#else
#define USE_ASYNC_COPY 0
#endif

typedef int v4i_vs __attribute__((vector_size(16)));

#if USE_ASYNC_COPY
#define ASYNC_B128(gsrc, ldst) \
  __builtin_amdgcn_global_load_async_to_lds_b128((v4i_vs*)(gsrc), (v4i_vs*)(ldst), 0, 0)
#define WAIT_ASYNC() __builtin_amdgcn_s_wait_asynccnt(0)
#else
#define ASYNC_B128(gsrc, ldst) \
  (*reinterpret_cast<uint4*>(ldst) = *reinterpret_cast<const uint4*>(gsrc))
#define WAIT_ASYNC() ((void)0)
#endif

// A-matrix 16x32 bf16 fragment (per ISA 7.12.2): lane m = lane&15;
// lanes 0-15 hold K {0..7,16..23}, lanes 16-31 hold K {8..15,24..31}.
__device__ __forceinline__ v16bf load_frag_a(const bf16_t* row32, int hi) {
  union { v16bf v; uint4 q[2]; } u;
  const uint4* p = reinterpret_cast<const uint4*>(row32);
  u.q[0] = p[hi];
  u.q[1] = p[2 + hi];
  return u.v;
}
// B-matrix 32x16 bf16 fragment: lane n = lane&15; lanes 0-15 hold K {0..7,16..23},
// lanes 16-31 hold K {8..15,24..31} (same k order as the A fragment).
__device__ __forceinline__ v16bf load_frag_b(const bf16_t* row32, int hi) {
  union { v16bf v; uint4 q[2]; } u;
  const uint4* p = reinterpret_cast<const uint4*>(row32);
  u.q[0] = p[hi];
  u.q[1] = p[2 + hi];
  return u.v;
}

// ---------------------------------------------------------------------------
// fp32 -> bf16 conversion
// ---------------------------------------------------------------------------
__global__ __launch_bounds__(256) void f32_to_bf16_kernel(
    const float* __restrict__ x, bf16_t* __restrict__ y, int n) {
  int i = blockIdx.x * 256 + threadIdx.x;
  if (i < n) y[i] = (bf16_t)x[i];
}

// ---------------------------------------------------------------------------
// Tiled WMMA GEMM:  C[M,N] = act(A[M,K] @ W[N,K]^T + bias [+ resid])
// Block tile 128x128, 8 waves (4x2), wave tile 32x64, K step 32.
// Double-buffered LDS, async global->LDS staging, one barrier per K step.
// ---------------------------------------------------------------------------
template <bool RELU, bool STORE_BF16, bool STORE_F32, bool ADD_RESID>
__global__ __launch_bounds__(256) void gemm_wmma_kernel(
    const bf16_t* __restrict__ A, const bf16_t* __restrict__ W,
    const float* __restrict__ bias, const float* __restrict__ resid,
    bf16_t* __restrict__ Cb, float* __restrict__ Cf, int M, int N, int K) {
  constexpr int BM = 128, BN = 128, KT = 32;
  __shared__ alignas(16) bf16_t As[2][BM][KT];
  __shared__ alignas(16) bf16_t Ws[2][BN][KT];

  const int tid = threadIdx.x;
  const int lane = tid & 31, wid = tid >> 5;
  const int wm = wid >> 1, wn = wid & 1;
  const int lm = lane & 15, hi = lane >> 4;
  const int m0 = blockIdx.y * BM, n0 = blockIdx.x * BN;

  // staging: 128 rows x 32 cols per operand; 256 threads x 2 rows x b128
  const int r0 = tid >> 2, c0 = (tid & 3) * 8;

  auto stage = [&](int buf, int k0) {
    ASYNC_B128(&A[(size_t)(m0 + r0) * K + k0 + c0],      &As[buf][r0][c0]);
    ASYNC_B128(&A[(size_t)(m0 + r0 + 64) * K + k0 + c0], &As[buf][r0 + 64][c0]);
    ASYNC_B128(&W[(size_t)(n0 + r0) * K + k0 + c0],      &Ws[buf][r0][c0]);
    ASYNC_B128(&W[(size_t)(n0 + r0 + 64) * K + k0 + c0], &Ws[buf][r0 + 64][c0]);
#if !USE_ASYNC_COPY
    if (k0 + KT < K) {  // global_prefetch_b8 on the synchronous path
      __builtin_prefetch(&A[(size_t)(m0 + r0) * K + k0 + KT + c0], 0, 1);
      __builtin_prefetch(&W[(size_t)(n0 + r0) * K + k0 + KT + c0], 0, 1);
    }
#endif
  };

  v8f acc[2][4] = {};

  stage(0, 0);
  for (int k0 = 0; k0 < K; k0 += KT) {
    const int buf = (k0 >> 5) & 1;
    WAIT_ASYNC();        // this wave's async fills of `buf` complete
    __syncthreads();     // all waves' fills visible; prior reads of buf^1 done
    if (k0 + KT < K) stage(buf ^ 1, k0 + KT);

    const v16bf af0 = load_frag_a(&As[buf][wm * 32 + lm][0], hi);
    const v16bf af1 = load_frag_a(&As[buf][wm * 32 + 16 + lm][0], hi);
#pragma unroll
    for (int j = 0; j < 4; ++j) {
      const v16bf wf = load_frag_b(&Ws[buf][wn * 64 + j * 16 + lm][0], hi);
      acc[0][j] = WMMA_BF16(af0, wf, acc[0][j]);
      acc[1][j] = WMMA_BF16(af1, wf, acc[1][j]);
    }
  }

  // C layout: lane holds col n = lane&15; VGPR r holds row r + 8*(lane>>4).
#pragma unroll
  for (int j = 0; j < 4; ++j) {
    const int n = n0 + wn * 64 + j * 16 + lm;
    const float bsc = bias[n];
#pragma unroll
    for (int t = 0; t < 2; ++t)
#pragma unroll
      for (int r = 0; r < 8; ++r) {
        const int m = m0 + wm * 32 + t * 16 + hi * 8 + r;
        float v = acc[t][j][r] + bsc;
        if constexpr (ADD_RESID) v += resid[(size_t)m * N + n];
        if constexpr (RELU) v = fmaxf(v, 0.f);
        if constexpr (STORE_F32) Cf[(size_t)m * N + n] = v;
        if constexpr (STORE_BF16) Cb[(size_t)m * N + n] = (bf16_t)v;
      }
  }
}

// ---------------------------------------------------------------------------
// Flash-style GQA attention. Grid: (S/64, NUM_HEADS, B). Block: 128 (4 waves).
// Each wave owns 16 query rows; the block streams 32-key tiles through LDS
// (K via async global->LDS, V transposed through VGPRs for the P@V B-frags).
// Output written in the reference's faithful view layout:
//   attnB[b][h*128 + 2*d + (s>>10)][s & 1023] = O[b,h,s,d]
// ---------------------------------------------------------------------------
__global__ __launch_bounds__(128) void attention_kernel(
    const bf16_t* __restrict__ Qb, const bf16_t* __restrict__ Kb,
    const bf16_t* __restrict__ Vb, bf16_t* __restrict__ attnB) {
  constexpr int S = 2048, D = 64, KV = 256, H = 1024;
  const int qt = blockIdx.x, h = blockIdx.y, b = blockIdx.z;
  const int g = h >> 2;  // consecutive-repeat KV group
  const int tid = threadIdx.x, lane = tid & 31, wid = tid >> 5;
  const int lm = lane & 15, hi = lane >> 4;
  const int s0 = qt * 64 + wid * 16;

  __shared__ alignas(16) bf16_t Ks[32][D];       // 32 keys x 64 dims
  __shared__ alignas(16) bf16_t VsT[D][40];      // transposed V tile (+pad)
  __shared__ alignas(16) bf16_t Ps[4][16][32];   // per-wave probabilities

  // Persistent Q fragments: A-matrix 16x32 over d (two halves of head_dim 64)
  const bf16_t* qrow = Qb + (size_t)(b * S + s0 + lm) * H + h * D;
  const v16bf qa0 = load_frag_a(qrow, hi);
  const v16bf qa1 = load_frag_a(qrow + 32, hi);

  float rmax[8], rsum[8];
  v8f o[4] = {};
#pragma unroll
  for (int r = 0; r < 8; ++r) { rmax[r] = -1e30f; rsum[r] = 0.f; }

  for (int kt = 0; kt < S; kt += 32) {
    __syncthreads();  // prior reads of Ks/VsT complete before refill
    const bf16_t* kb = Kb + (size_t)(b * S + kt) * KV + g * D;
    const bf16_t* vb = Vb + (size_t)(b * S + kt) * KV + g * D;
#pragma unroll
    for (int i = tid; i < 256; i += 128) {
      const int row = i >> 3, c = (i & 7) * 8;
      ASYNC_B128(&kb[(size_t)row * KV + c], &Ks[row][c]);
      union { uint4 q; bf16_t e[8]; } u;
      u.q = *reinterpret_cast<const uint4*>(&vb[(size_t)row * KV + c]);
#pragma unroll
      for (int e = 0; e < 8; ++e) VsT[c + e][row] = u.e[e];  // transpose for P@V
    }
    WAIT_ASYNC();
    __syncthreads();

    // S = Q @ K^T : two 16x16 score tiles (32 keys), K-dim = head_dim
    v8f sc[2];
#pragma unroll
    for (int t = 0; t < 2; ++t) {
      v8f sv = {};
      const bf16_t* krow = &Ks[t * 16 + lm][0];
      sv = WMMA_BF16(qa0, load_frag_b(krow, hi), sv);
      sv = WMMA_BF16(qa1, load_frag_b(krow + 32, hi), sv);
      sc[t] = sv;
    }

    // online softmax over the 32 new keys (row = r + 8*hi, spread over 16 lanes)
    float tmax[8];
#pragma unroll
    for (int r = 0; r < 8; ++r) {
      const float a = sc[0][r] * 0.125f;  // 1/sqrt(64)
      const float c = sc[1][r] * 0.125f;
      sc[0][r] = a; sc[1][r] = c;
      tmax[r] = fmaxf(a, c);
    }
#pragma unroll
    for (int off = 1; off < 16; off <<= 1)
#pragma unroll
      for (int r = 0; r < 8; ++r) tmax[r] = fmaxf(tmax[r], __shfl_xor(tmax[r], off, 32));

    float alpha[8], psum[8];
#pragma unroll
    for (int r = 0; r < 8; ++r) {
      const float mn = fmaxf(rmax[r], tmax[r]);
      alpha[r] = __expf(rmax[r] - mn);
      rmax[r] = mn;
      psum[r] = 0.f;
    }
#pragma unroll
    for (int t = 0; t < 2; ++t)
#pragma unroll
      for (int r = 0; r < 8; ++r) {
        const float p = __expf(sc[t][r] - rmax[r]);
        psum[r] += p;
        Ps[wid][hi * 8 + r][t * 16 + lm] = (bf16_t)p;  // wave-private restage
      }
#pragma unroll
    for (int off = 1; off < 16; off <<= 1)
#pragma unroll
      for (int r = 0; r < 8; ++r) psum[r] += __shfl_xor(psum[r], off, 32);
#pragma unroll
    for (int r = 0; r < 8; ++r) rsum[r] = rsum[r] * alpha[r] + psum[r];
#pragma unroll
    for (int j = 0; j < 4; ++j)
#pragma unroll
      for (int r = 0; r < 8; ++r) o[j][r] *= alpha[r];

    // O += P @ V : K-dim = 32 keys, four 16-wide d sub-tiles
    const v16bf pf = load_frag_a(&Ps[wid][lm][0], hi);
#pragma unroll
    for (int j = 0; j < 4; ++j) {
      const v16bf vf = load_frag_b(&VsT[j * 16 + lm][0], hi);
      o[j] = WMMA_BF16(pf, vf, o[j]);
    }
  }

  // normalize and scatter into the faithful flattened-view buffer
#pragma unroll
  for (int j = 0; j < 4; ++j) {
    const int d = j * 16 + lm;
#pragma unroll
    for (int r = 0; r < 8; ++r) {
      const int s = s0 + hi * 8 + r;
      const float val = o[j][r] / rsum[r];
      const int rr = h * 128 + d * 2 + (s >> 10);
      attnB[((size_t)b * 2048 + rr) * 1024 + (s & 1023)] = (bf16_t)val;
    }
  }
}

// ---------------------------------------------------------------------------
// LayerNorm over rows of length H=1024. One block (256 threads) per row.
// In-place safe (all reads complete before the barrier preceding writes).
// ---------------------------------------------------------------------------
template <bool ADD_RESID, bool WRITE_F32, bool WRITE_BF16>
__global__ __launch_bounds__(256) void layernorm_kernel(
    const float* __restrict__ X, const float* __restrict__ R,
    const float* __restrict__ gamma, const float* __restrict__ beta,
    float* __restrict__ outF, bf16_t* __restrict__ outB, int H) {
  const int row = blockIdx.x, tid = threadIdx.x;
  const int lane = tid & 31, wid = tid >> 5;
  const float* x = X + (size_t)row * H;
  float v[4], s = 0.f, sq = 0.f;
#pragma unroll
  for (int i = 0; i < 4; ++i) {
    const int c = tid + i * 256;
    float t = x[c];
    if constexpr (ADD_RESID) t += R[(size_t)row * H + c];
    v[i] = t; s += t; sq += t * t;
  }
#pragma unroll
  for (int off = 16; off; off >>= 1) {
    s += __shfl_xor(s, off, 32);
    sq += __shfl_xor(sq, off, 32);
  }
  __shared__ float ps[8], pq[8];
  if (lane == 0) { ps[wid] = s; pq[wid] = sq; }
  __syncthreads();
  float S = 0.f, Q = 0.f;
#pragma unroll
  for (int i = 0; i < 8; ++i) { S += ps[i]; Q += pq[i]; }
  const float mu = S / (float)H;
  const float inv = rsqrtf(fmaxf(Q / (float)H - mu * mu, 0.f) + 1e-5f);
#pragma unroll
  for (int i = 0; i < 4; ++i) {
    const int c = tid + i * 256;
    const float y = (v[i] - mu) * inv * gamma[c] + beta[c];
    if constexpr (WRITE_F32) outF[(size_t)row * H + c] = y;
    if constexpr (WRITE_BF16) outB[(size_t)row * H + c] = (bf16_t)y;
  }
}

// ---------------------------------------------------------------------------
// Host orchestration. Workspace layout (bytes), ~69 MB total:
//   [0,8M)   Xb (hidden bf16)        [8M,17M)  weights bf16 (wq,wk,wv,wo,w1,w2)
//   [17M,25M) Qb  [25M,27M) Kb  [27M,29M) Vb   [29M,37M) attn view bf16
//   [37M,53M) pre/post-LN f32        [53M,61M) LN out bf16
//   [61M,69M) relu(ffn1) bf16        ffn2 f32 reuses [17M,33M)
// ---------------------------------------------------------------------------
extern "C" void kernel_launch(void* const* d_in, const int* in_sizes, int n_in,
                              void* d_out, int out_size, void* d_ws, size_t ws_size,
                              hipStream_t stream) {
  (void)in_sizes; (void)n_in; (void)out_size; (void)ws_size;
  const float* hidden = (const float*)d_in[0];
  const float* wq = (const float*)d_in[1];
  const float* bq = (const float*)d_in[2];
  const float* wk = (const float*)d_in[3];
  const float* bk = (const float*)d_in[4];
  const float* wv = (const float*)d_in[5];
  const float* bvv = (const float*)d_in[6];
  const float* wo = (const float*)d_in[7];
  const float* bo = (const float*)d_in[8];
  const float* ln_g = (const float*)d_in[9];
  const float* ln_b = (const float*)d_in[10];
  const float* w1 = (const float*)d_in[11];
  const float* b1 = (const float*)d_in[12];
  const float* w2 = (const float*)d_in[13];
  const float* b2 = (const float*)d_in[14];
  const float* fg = (const float*)d_in[15];
  const float* fb = (const float*)d_in[16];
  float* out = (float*)d_out;

  const size_t MB = 1ull << 20;
  char* w = (char*)d_ws;
  bf16_t* Xb   = (bf16_t*)(w + 0);
  bf16_t* WqB  = (bf16_t*)(w + 8 * MB);
  bf16_t* WkB  = (bf16_t*)(w + 10 * MB);
  bf16_t* WvB  = (bf16_t*)(w + 10 * MB + 512 * 1024);
  bf16_t* WoB  = (bf16_t*)(w + 11 * MB);
  bf16_t* W1B  = (bf16_t*)(w + 13 * MB);
  bf16_t* W2B  = (bf16_t*)(w + 15 * MB);
  bf16_t* Qb   = (bf16_t*)(w + 17 * MB);
  bf16_t* Kb   = (bf16_t*)(w + 25 * MB);
  bf16_t* Vb   = (bf16_t*)(w + 27 * MB);
  bf16_t* AtB  = (bf16_t*)(w + 29 * MB);
  float*  preF = (float*) (w + 37 * MB);
  bf16_t* outB = (bf16_t*)(w + 53 * MB);
  bf16_t* F1B  = (bf16_t*)(w + 61 * MB);
  float*  F2F  = (float*) (w + 17 * MB);  // reuse Q/K/V region after attention

  const int M = 4096, H = 1024, KV = 256;
  const dim3 blk(256);

  auto cvt = [&](const float* src, bf16_t* dst, int n) {
    f32_to_bf16_kernel<<<(n + 255) / 256, 256, 0, stream>>>(src, dst, n);
  };
  cvt(hidden, Xb, M * H);
  cvt(wq, WqB, H * H);
  cvt(wk, WkB, KV * H);
  cvt(wv, WvB, KV * H);
  cvt(wo, WoB, H * H);
  cvt(w1, W1B, H * H);
  cvt(w2, W2B, H * H);

  // QKV projections (bf16 outputs); block tile 128x128
  gemm_wmma_kernel<false, true, false, false>
      <<<dim3(H / 128, M / 128), blk, 0, stream>>>(Xb, WqB, bq, nullptr, Qb, nullptr, M, H, H);
  gemm_wmma_kernel<false, true, false, false>
      <<<dim3(KV / 128, M / 128), blk, 0, stream>>>(Xb, WkB, bk, nullptr, Kb, nullptr, M, KV, H);
  gemm_wmma_kernel<false, true, false, false>
      <<<dim3(KV / 128, M / 128), blk, 0, stream>>>(Xb, WvB, bvv, nullptr, Vb, nullptr, M, KV, H);

  // Flash attention with GQA group sharing
  attention_kernel<<<dim3(2048 / 64, 16, 2), dim3(128), 0, stream>>>(Qb, Kb, Vb, AtB);

  // out-projection + fp32 residual, then LN (in place) -> preF (f32) + outB (bf16)
  gemm_wmma_kernel<false, false, true, true>
      <<<dim3(H / 128, M / 128), blk, 0, stream>>>(AtB, WoB, bo, hidden, nullptr, preF, M, H, H);
  layernorm_kernel<false, true, true>
      <<<dim3(M), blk, 0, stream>>>(preF, nullptr, ln_g, ln_b, preF, outB, H);

  // FFN: relu(out@w1^T+b1) -> F1B ; F1B@w2^T+b2 -> F2F ; final LN(out + ffn)
  gemm_wmma_kernel<true, true, false, false>
      <<<dim3(H / 128, M / 128), blk, 0, stream>>>(outB, W1B, b1, nullptr, F1B, nullptr, M, H, H);
  gemm_wmma_kernel<false, false, true, false>
      <<<dim3(H / 128, M / 128), blk, 0, stream>>>(F1B, W2B, b2, nullptr, nullptr, F2F, M, H, H);
  layernorm_kernel<true, true, false>
      <<<dim3(M), blk, 0, stream>>>(preF, F2F, fg, fb, out, nullptr, H);
}
